// EMACodebook_51608327028991
// MI455X (gfx1250) — compile-verified, matrix operand order checked by reading, not run on hardware
//
#include <hip/hip_runtime.h>
#include <cstdint>

typedef __attribute__((ext_vector_type(16))) __bf16 v16bf;
typedef __attribute__((ext_vector_type(8)))  float  v8f;

#define CB_SIZE 2048
#define DIM     64
#define BATCH   32
#define FRAMES  4096
#define NTOT    (BATCH*FRAMES)      // 131072 rows
#define DECAYF  0.99f
#define OMD     0.01f
#define EPSF    1e-5f

// ---- workspace layout (32-bit words) ----
#define OFF_EHI    0                 // 65536 u32  : packed bf16-hi of E (B-fragment order)
#define OFF_ELO    65536             // 65536 u32  : packed bf16-lo of E
#define OFF_ENORM  131072            // 2048 f32   : ||e||^2
#define OFF_HIST   133120            // 2048 f32   : cs_batch accumulator
#define OFF_EMBSUM 135168            // 131072 f32 : emb_sum accumulator
#define OFF_LOSS   266240            // 1 f32
#define OFF_N      266241            // 1 f32
#define OFF_NCS    266242            // 2048 f32   : new_cluster_size copy
#define ZERO_START OFF_HIST
#define ZERO_COUNT (OFF_NCS + CB_SIZE - OFF_HIST)   // 135170

// ---- d_out layout (floats, tuple order) ----
#define OUT_Q    0
#define OUT_IDX  8388608
#define OUT_LOSS 8519680
#define OUT_EMB  8519681
#define OUT_NCS  8650753
#define OUT_AVG  8652801

__device__ __forceinline__ uint32_t f2bf(float x) {
  uint32_t b = __float_as_uint(x);
  return (b + 0x7FFFu + ((b >> 16) & 1u)) >> 16;   // RNE truncate to bf16
}

union FragU {
  uint32_t u[8];
  uint4    q[2];
  v16bf    h;
};

// ---------------- zero accumulators ----------------
__global__ void k_zero(float* __restrict__ ws) {
  int i = blockIdx.x * blockDim.x + threadIdx.x;
  if (i < ZERO_COUNT) ws[ZERO_START + i] = 0.0f;
}

// ---------------- pack E into WMMA B-fragment order, split hi/lo bf16 ----------------
// index q = ((t*2 + f)*32 + lane); each q owns 8 consecutive u32 (one per B vgpr).
__global__ void k_pack(const float* __restrict__ emb,
                       uint32_t* __restrict__ ehi, uint32_t* __restrict__ elo) {
  int q = blockIdx.x * blockDim.x + threadIdx.x;
  if (q >= (CB_SIZE / 16) * 2 * 32) return;
  int lane = q & 31;
  int f    = (q >> 5) & 1;          // K fragment: 0 -> K 0..31, 1 -> K 32..63
  int t    = q >> 6;                // code tile
  int j    = t * 16 + (lane & 15);  // code index (N position = lane%16)
  int half = lane >> 4;
#pragma unroll
  for (int v = 0; v < 8; ++v) {
    int k = f * 32 + ((v < 4) ? (half * 8 + 2 * v) : (16 + half * 8 + 2 * (v - 4)));
    float e0 = emb[j * DIM + k];
    float e1 = emb[j * DIM + k + 1];
    uint32_t h0 = f2bf(e0), h1 = f2bf(e1);
    float r0 = e0 - __uint_as_float(h0 << 16);
    float r1 = e1 - __uint_as_float(h1 << 16);
    ehi[q * 8 + v] = h0 | (h1 << 16);
    elo[q * 8 + v] = f2bf(r0) | (f2bf(r1) << 16);
  }
}

// ---------------- ||e||^2 ----------------
__global__ void k_norm(const float* __restrict__ emb, float* __restrict__ enorm) {
  int j = blockIdx.x * blockDim.x + threadIdx.x;
  if (j >= CB_SIZE) return;
  float s = 0.0f;
#pragma unroll 8
  for (int c = 0; c < DIM; ++c) { float e = emb[j * DIM + c]; s += e * e; }
  enorm[j] = s;
}

// ---------------- main: WMMA nearest-code search + outputs + scatters ----------------
__global__ __launch_bounds__(256) void k_assign(
    const float* __restrict__ in, const float* __restrict__ emb,
    const uint32_t* __restrict__ ehi, const uint32_t* __restrict__ elo,
    const float* __restrict__ enorm,
    float* __restrict__ out_q, float* __restrict__ out_idx,
    float* __restrict__ hist, float* __restrict__ embsum, float* __restrict__ loss) {
  __shared__ uint32_t sXhi[128 * 33];   // padded stride (33) to avoid bank conflicts
  __shared__ uint32_t sXlo[128 * 33];
  __shared__ int      sIdx[128];
  __shared__ float    sRed[256];

  const int  tid   = threadIdx.x;
  const int  g     = blockIdx.x;
  const int  n0    = g * 128;           // first flat row of this tile
  const int  b     = n0 >> 12;          // n = b*4096 + f
  const int  f0    = n0 & 4095;
  const long bbase = (long)b * DIM * FRAMES;

  // ---- phase A: stage 128x64 X tile into LDS as packed (hi,lo) bf16 pairs ----
#pragma unroll
  for (int i = 0; i < 16; ++i) {
    int ii = i * 256 + tid;             // 128 rows x 32 c-pairs
    int cp = ii >> 7;
    int r  = ii & 127;
    float x0 = in[bbase + (long)(2 * cp)     * FRAMES + f0 + r];
    float x1 = in[bbase + (long)(2 * cp + 1) * FRAMES + f0 + r];
    uint32_t h0 = f2bf(x0), h1 = f2bf(x1);
    float l0 = x0 - __uint_as_float(h0 << 16);
    float l1 = x1 - __uint_as_float(h1 << 16);
    sXhi[r * 33 + cp] = h0 | (h1 << 16);
    sXlo[r * 33 + cp] = f2bf(l0) | (f2bf(l1) << 16);
  }
  __syncthreads();

  const int lane = tid & 31;
  const int w    = tid >> 5;            // wave id: owns rows 16w..16w+15
  const int half = lane >> 4;
  const int mrow = w * 16 + (lane & 15);

  // ---- build A fragments once (ISA 16-bit A 16x32 layout) ----
  FragU aH[2], aL[2];
#pragma unroll
  for (int f = 0; f < 2; ++f)
#pragma unroll
    for (int v = 0; v < 8; ++v) {
      int col = f * 16 + ((v < 4) ? (half * 4 + v) : (8 + half * 4 + (v - 4)));
      aH[f].u[v] = sXhi[mrow * 33 + col];
      aL[f].u[v] = sXlo[mrow * 33 + col];
    }

  const uint4* ehi4 = (const uint4*)ehi;
  const uint4* elo4 = (const uint4*)elo;

  float bestD[8];
  int   bestJ[8];
#pragma unroll
  for (int r = 0; r < 8; ++r) { bestD[r] = 3.4e38f; bestJ[r] = 0; }

  // ---- sweep all 128 code tiles ----
  for (int t = 0; t < CB_SIZE / 16; ++t) {
    FragU bH0, bH1, bL0, bL1;
    int q0 = ((t * 2 + 0) * 32 + lane) * 2;
    int q1 = ((t * 2 + 1) * 32 + lane) * 2;
    bH0.q[0] = ehi4[q0]; bH0.q[1] = ehi4[q0 + 1];
    bH1.q[0] = ehi4[q1]; bH1.q[1] = ehi4[q1 + 1];
    bL0.q[0] = elo4[q0]; bL0.q[1] = elo4[q0 + 1];
    bL1.q[0] = elo4[q1]; bL1.q[1] = elo4[q1 + 1];
    float en = enorm[t * 16 + (lane & 15)];

    v8f acc = {};
    // x_hi.e_hi + x_lo.e_hi + x_hi.e_lo  (split-precision bf16 ~= fp32 dot)
    acc = __builtin_amdgcn_wmma_f32_16x16x32_bf16(false, aH[0].h, false, bH0.h, (short)0, acc, false, false);
    acc = __builtin_amdgcn_wmma_f32_16x16x32_bf16(false, aH[1].h, false, bH1.h, (short)0, acc, false, false);
    acc = __builtin_amdgcn_wmma_f32_16x16x32_bf16(false, aL[0].h, false, bH0.h, (short)0, acc, false, false);
    acc = __builtin_amdgcn_wmma_f32_16x16x32_bf16(false, aL[1].h, false, bH1.h, (short)0, acc, false, false);
    acc = __builtin_amdgcn_wmma_f32_16x16x32_bf16(false, aH[0].h, false, bL0.h, (short)0, acc, false, false);
    acc = __builtin_amdgcn_wmma_f32_16x16x32_bf16(false, aH[1].h, false, bL1.h, (short)0, acc, false, false);

    int j = t * 16 + (lane & 15);
#pragma unroll
    for (int r = 0; r < 8; ++r) {
      float cand = en - 2.0f * acc[r];  // ||e||^2 - 2 x.e (||x||^2 row-constant)
      if (cand < bestD[r]) { bestD[r] = cand; bestJ[r] = j; }
    }
  }

  // ---- cross-lane argmin within groups of 16 (C layout: lane = N, vgpr r = M) ----
#pragma unroll
  for (int m = 1; m <= 8; m <<= 1) {
#pragma unroll
    for (int r = 0; r < 8; ++r) {
      float od = __shfl_xor(bestD[r], m);
      int   oj = __shfl_xor(bestJ[r], m);
      if (od < bestD[r] || (od == bestD[r] && oj < bestJ[r])) { bestD[r] = od; bestJ[r] = oj; }
    }
  }
  if ((lane & 15) == 0) {
#pragma unroll
    for (int r = 0; r < 8; ++r) sIdx[w * 16 + half * 8 + r] = bestJ[r];
  }
  __syncthreads();

  // ---- phase 2: outputs + histogram + scatter + loss ----
  if (tid < 128) {
    int j = sIdx[tid];
    out_idx[n0 + tid] = (float)j;
    atomicAdd(&hist[j], 1.0f);
  }

  float lsum = 0.0f;
  for (int i = 0; i < 32; ++i) {
    int ii = i * 256 + tid;             // 128 rows x 64 channels
    int c  = ii >> 7;
    int r  = ii & 127;
    int j  = sIdx[r];
    float ev = emb[j * DIM + c];
    long  ga = bbase + (long)c * FRAMES + f0 + r;
    float xv = in[ga];
    out_q[ga] = ev;                     // STE forward value == embedding[idx]
    float d = xv - ev;
    lsum += d * d;
    atomicAdd(&embsum[j * DIM + c], xv);
  }
  sRed[tid] = lsum;
  __syncthreads();
  for (int s = 128; s > 0; s >>= 1) {
    if (tid < s) sRed[tid] += sRed[tid + s];
    __syncthreads();
  }
  if (tid == 0) atomicAdd(loss, sRed[0]);
}

// ---------------- finalize: cluster sizes, n, loss ----------------
__global__ __launch_bounds__(256) void k_fin1(const float* __restrict__ cs,
                                              float* __restrict__ ws,
                                              float* __restrict__ out) {
  __shared__ float sRed[256];
  int tid = threadIdx.x;
  float sum = 0.0f;
  for (int j = tid; j < CB_SIZE; j += 256) {
    float ncs = DECAYF * cs[j] + OMD * ws[OFF_HIST + j];
    out[OUT_NCS + j] = ncs;
    ws[OFF_NCS + j]  = ncs;
    sum += ncs;
  }
  sRed[tid] = sum;
  __syncthreads();
  for (int s = 128; s > 0; s >>= 1) {
    if (tid < s) sRed[tid] += sRed[tid + s];
    __syncthreads();
  }
  if (tid == 0) {
    ws[OFF_N] = sRed[0];
    out[OUT_LOSS] = ws[OFF_LOSS] / (float)(NTOT * DIM);
  }
}

// ---------------- finalize: EMA avg + normalized embedding ----------------
__global__ void k_fin2(const float* __restrict__ avg,
                       const float* __restrict__ ws,
                       float* __restrict__ out) {
  int i = blockIdx.x * blockDim.x + threadIdx.x;
  if (i >= CB_SIZE * DIM) return;
  int j = i >> 6;
  float navg = DECAYF * avg[i] + OMD * ws[OFF_EMBSUM + i];
  out[OUT_AVG + i] = navg;
  float n   = ws[OFF_N];
  float ncs = ws[OFF_NCS + j];
  float smoothed = (ncs + EPSF) / (n + (float)CB_SIZE * EPSF) * fmaxf(n, 1.0f);
  out[OUT_EMB + i] = navg / smoothed;
}

extern "C" void kernel_launch(void* const* d_in, const int* in_sizes, int n_in,
                              void* d_out, int out_size, void* d_ws, size_t ws_size,
                              hipStream_t stream) {
  const float* in  = (const float*)d_in[0];
  const float* emb = (const float*)d_in[1];
  const float* avg = (const float*)d_in[2];
  const float* cs  = (const float*)d_in[3];
  float*    out = (float*)d_out;
  float*    ws  = (float*)d_ws;
  uint32_t* wsu = (uint32_t*)d_ws;

  k_zero<<<(ZERO_COUNT + 255) / 256, 256, 0, stream>>>(ws);
  k_pack<<<(8192 + 255) / 256, 256, 0, stream>>>(emb, wsu + OFF_EHI, wsu + OFF_ELO);
  k_norm<<<(CB_SIZE + 255) / 256, 256, 0, stream>>>(emb, ws + OFF_ENORM);
  k_assign<<<NTOT / 128, 256, 0, stream>>>(in, emb, wsu + OFF_EHI, wsu + OFF_ELO,
                                           ws + OFF_ENORM,
                                           out + OUT_Q, out + OUT_IDX,
                                           ws + OFF_HIST, ws + OFF_EMBSUM, ws + OFF_LOSS);
  k_fin1<<<1, 256, 0, stream>>>(cs, ws, out);
  k_fin2<<<(CB_SIZE * DIM + 255) / 256, 256, 0, stream>>>(avg, ws, out);
}
